// S3FD_57269093925070
// MI455X (gfx1250) — matrix operand order repaired and checked by reading, not compiled, not run on hardware
//
#include <hip/hip_runtime.h>
#include <hip/hip_bf16.h>
#include <math.h>

typedef __attribute__((ext_vector_type(16))) _Float16 v16h;
typedef __attribute__((ext_vector_type(8)))  float    v8f;

#define BM 64
#define BN 128
#define BK 32

// ---------------------------------------------------------------------------
// Implicit-GEMM convolution. GEMM: M = Cout, N = Hout*Wout, K = Cin*KH*KW.
// Block = 256 threads (8 waves, wave32) arranged 2(Mwave) x 4(Nwave).
// Block tile 64(M) x 128(N); each wave computes a 2x2 grid of 16x16 C tiles
// => 4 x v_wmma_f32_16x16x32_f16 per K-step with 2 A-frag + 2 B-frag loads.
// LDS tiles are stored in WMMA *fragment order* so each fragment load is one
// contiguous 32-byte (16 x f16) read per lane.
//
// Fragment-order mapping (per 16x16(x32) tile, contiguous k, same in A and B):
//  A (16x32, f16): lane = (m&15) + 16*(k>=16),     slot = k%16
//  B (32x16, f16): lane = (n&15) + 16*(k>=16),     slot = k%16
//  C (16x16, f32): VGPR r -> m = r + 8*(lane>=16), n = lane&15
// ---------------------------------------------------------------------------
__global__ __launch_bounds__(256) void conv_wmma_kernel(
    const float* __restrict__ X, const float* __restrict__ Wt,
    const float* __restrict__ bias, float* __restrict__ Y,
    int Cin, int Cout, int Hin, int Win, int Hout, int Wout,
    int KH, int KW, int stride, int pad, int dil, int relu)
{
    const int K    = Cin * KH * KW;
    const int Npix = Hout * Wout;
    const int m0   = blockIdx.y * BM;
    const int n0   = blockIdx.x * BN;

    // fragment-order staging: [tile][lane][slot]
    __shared__ __align__(32) _Float16 Af[4][32][16];   // 4 M-tiles
    __shared__ __align__(32) _Float16 Bf[8][32][16];   // 8 N-tiles

    const int tid  = threadIdx.x;
    const int wave = tid >> 5;
    const int lane = tid & 31;
    const int wr   = wave >> 2;   // 0..1 : wave row (M)
    const int wc   = wave & 3;    // 0..3 : wave col (N)

    v8f acc00 = {0.f,0.f,0.f,0.f,0.f,0.f,0.f,0.f};
    v8f acc01 = acc00, acc10 = acc00, acc11 = acc00;

    for (int k0 = 0; k0 < K; k0 += BK) {
        // prefetch next weight tile (global_prefetch_b8 on gfx1250)
        __builtin_prefetch((const void*)(Wt + (size_t)m0 * K + k0 + BK), 0, 1);

        // --- stage A (weights, contiguous in K for OIHW) into fragment order ---
        for (int i = tid; i < BM * BK; i += 256) {
            int m = i >> 5;          // / BK
            int k = i & 31;
            int gm = m0 + m, gk = k0 + k;
            float v = 0.f;
            if (gm < Cout && gk < K) v = Wt[(size_t)gm * K + gk];
            int hi     = k >> 4;         // k >= 16
            int kk     = k & 15;
            int lanehi = kk >> 3;        // kk >= 8
            Af[m >> 4][(m & 15) + 16 * hi][kk] = (_Float16)v;
        }
        // --- stage B (im2col gather) into fragment order ---
        for (int i = tid; i < BK * BN; i += 256) {
            int k = i >> 7;          // / BN
            int n = i & 127;
            int gk = k0 + k, gn = n0 + n;
            float v = 0.f;
            if (gk < K && gn < Npix) {
                int ci = gk / (KH * KW);
                int r  = gk % (KH * KW);
                int ky = r / KW, kx = r % KW;
                int oy = gn / Wout, ox = gn % Wout;
                int iy = oy * stride - pad + ky * dil;
                int ix = ox * stride - pad + kx * dil;
                if (iy >= 0 && iy < Hin && ix >= 0 && ix < Win)
                    v = X[((size_t)ci * Hin + iy) * Win + ix];
            }
            Bf[n >> 4][(n & 15) + 16 * (k >> 4)][k & 15] = (_Float16)v;
        }
        __syncthreads();

        // --- contiguous 32B fragment loads + 2x2 WMMA ---
        v16h a0 = *(const v16h*)&Af[2 * wr + 0][lane][0];
        v16h a1 = *(const v16h*)&Af[2 * wr + 1][lane][0];
        v16h b0 = *(const v16h*)&Bf[2 * wc + 0][lane][0];
        v16h b1 = *(const v16h*)&Bf[2 * wc + 1][lane][0];

        acc00 = __builtin_amdgcn_wmma_f32_16x16x32_f16(false, a0, false, b0, (short)0, acc00, false, false);
        acc01 = __builtin_amdgcn_wmma_f32_16x16x32_f16(false, a0, false, b1, (short)0, acc01, false, false);
        acc10 = __builtin_amdgcn_wmma_f32_16x16x32_f16(false, a1, false, b0, (short)0, acc10, false, false);
        acc11 = __builtin_amdgcn_wmma_f32_16x16x32_f16(false, a1, false, b1, (short)0, acc11, false, false);
        __syncthreads();
    }

    // --- store 2x2 tiles of C ---
    const int nlo = lane & 15;
    const int moff = (lane >> 4) * 8;
    #pragma unroll
    for (int ta = 0; ta < 2; ++ta) {
        #pragma unroll
        for (int tb = 0; tb < 2; ++tb) {
            v8f acc = (ta == 0) ? ((tb == 0) ? acc00 : acc01)
                                : ((tb == 0) ? acc10 : acc11);
            int mt = m0 + (2 * wr + ta) * 16 + moff;
            int nn = n0 + (2 * wc + tb) * 16 + nlo;
            #pragma unroll
            for (int r = 0; r < 8; ++r) {
                int m = mt + r;
                if (m < Cout && nn < Npix) {
                    float v = acc[r] + bias[m];
                    if (relu) v = fmaxf(v, 0.f);
                    Y[(size_t)m * Npix + nn] = v;
                }
            }
        }
    }
}

// ---------------------------------------------------------------------------
// 2x2 max pool, stride 2
// ---------------------------------------------------------------------------
__global__ void pool_kernel(const float* __restrict__ X, float* __restrict__ Y,
                            int C, int Hin, int Win)
{
    int Hout = Hin >> 1, Wout = Win >> 1;
    int idx = blockIdx.x * blockDim.x + threadIdx.x;
    int total = C * Hout * Wout;
    if (idx >= total) return;
    int ox = idx % Wout;
    int t  = idx / Wout;
    int oy = t % Hout;
    int c  = t / Hout;
    const float* p = X + ((size_t)c * Hin + 2 * oy) * Win + 2 * ox;
    Y[idx] = fmaxf(fmaxf(p[0], p[1]), fmaxf(p[Win], p[Win + 1]));
}

// ---------------------------------------------------------------------------
// Channel-wise L2 norm with learned gain
// ---------------------------------------------------------------------------
__global__ void l2norm_kernel(const float* __restrict__ X, const float* __restrict__ g,
                              float* __restrict__ Y, int C, int HW)
{
    int idx = blockIdx.x * blockDim.x + threadIdx.x;
    if (idx >= HW) return;
    float s = 0.f;
    for (int c = 0; c < C; ++c) { float v = X[(size_t)c * HW + idx]; s += v * v; }
    float inv = 1.0f / (sqrtf(s) + 1e-10f);
    for (int c = 0; c < C; ++c)
        Y[(size_t)c * HW + idx] = g[c] * X[(size_t)c * HW + idx] * inv;
}

// ---------------------------------------------------------------------------
// Per-scale decode: maxout (scale 0), softmax over 2 classes, prior decode
// ---------------------------------------------------------------------------
__global__ void decode_kernel(const float* __restrict__ loc, const float* __restrict__ conf,
                              float* __restrict__ boxes, float* __restrict__ scores,
                              int f, int step, float msize, int base, int confC)
{
    int idx = blockIdx.x * blockDim.x + threadIdx.x;
    int HW = f * f;
    if (idx >= HW) return;
    int i = idx / f, j = idx % f;
    float cx = (j + 0.5f) * step / 640.f;
    float cy = (i + 0.5f) * step / 640.f;
    float s  = msize / 640.f;

    float l0 = loc[0 * HW + idx], l1 = loc[1 * HW + idx];
    float l2 = loc[2 * HW + idx], l3 = loc[3 * HW + idx];

    float c0, c1;
    if (confC == 4) {
        c0 = fmaxf(fmaxf(conf[0 * HW + idx], conf[1 * HW + idx]), conf[2 * HW + idx]);
        c1 = conf[3 * HW + idx];
    } else {
        c0 = conf[0 * HW + idx];
        c1 = conf[1 * HW + idx];
    }
    float mx = fmaxf(c0, c1);
    float e0 = __expf(c0 - mx), e1 = __expf(c1 - mx);
    float score = e1 / (e0 + e1);

    float bx = cx + l0 * 0.1f * s;
    float by = cy + l1 * 0.1f * s;
    float bw = s * __expf(l2 * 0.2f);
    float bh = s * __expf(l3 * 0.2f);
    float x1 = bx - 0.5f * bw, y1 = by - 0.5f * bh;

    int g = base + idx;
    boxes[4 * g + 0] = x1;
    boxes[4 * g + 1] = y1;
    boxes[4 * g + 2] = x1 + bw;
    boxes[4 * g + 3] = y1 + bh;
    scores[g] = score;
}

// ---------------------------------------------------------------------------
// Top-K (K=750) by iterative argmax over masked scores; single block
// ---------------------------------------------------------------------------
__global__ __launch_bounds__(1024) void topk_kernel(
    const float* __restrict__ scores, int* __restrict__ taken, int N,
    float* __restrict__ topi_out, float* __restrict__ tops_out, int K)
{
    __shared__ float sv[1024];
    __shared__ int   si[1024];
    const int tid = threadIdx.x;
    for (int i = tid; i < N; i += 1024) taken[i] = 0;
    __syncthreads();

    for (int k = 0; k < K; ++k) {
        float bv = -__builtin_inff();
        int   bi = N;
        for (int i = tid; i < N; i += 1024) {
            if (taken[i]) continue;
            float sc = scores[i];
            float v  = (sc > 0.5f) ? sc : -__builtin_inff();
            if (v > bv || (v == bv && i < bi)) { bv = v; bi = i; }
        }
        sv[tid] = bv; si[tid] = bi;
        __syncthreads();
        for (int w = 512; w > 0; w >>= 1) {
            if (tid < w) {
                float ov = sv[tid + w]; int oi = si[tid + w];
                if (ov > sv[tid] || (ov == sv[tid] && oi < si[tid])) {
                    sv[tid] = ov; si[tid] = oi;
                }
            }
            __syncthreads();
        }
        if (tid == 0) {
            int sel = (si[0] < N) ? si[0] : 0;
            topi_out[k] = (float)sel;
            tops_out[k] = sv[0];
            taken[sel]  = 1;
        }
        __syncthreads();
    }
}

// ---------------------------------------------------------------------------
// Sequential greedy NMS over the top-750, single block
// ---------------------------------------------------------------------------
#define NMS_K 750
__global__ __launch_bounds__(768) void nms_kernel(
    const float* __restrict__ boxes, const float* __restrict__ topi_f,
    const float* __restrict__ tops, float* __restrict__ keep_out)
{
    __shared__ float x1[NMS_K], y1[NMS_K], x2[NMS_K], y2[NMS_K], area[NMS_K];
    __shared__ int supp[NMS_K];
    const int tid = threadIdx.x;
    if (tid < NMS_K) {
        int bi = (int)topi_f[tid];
        float a = boxes[4 * bi + 0], b = boxes[4 * bi + 1];
        float c = boxes[4 * bi + 2], d = boxes[4 * bi + 3];
        x1[tid] = a; y1[tid] = b; x2[tid] = c; y2[tid] = d;
        area[tid] = (c - a) * (d - b);
        supp[tid] = 0;
    }
    __syncthreads();
    for (int i = 0; i < NMS_K; ++i) {
        bool keep_i = (tops[i] > -1e37f) && (supp[i] == 0);
        if (tid < NMS_K && tid > i && keep_i) {
            float iw = fmaxf(fminf(x2[i], x2[tid]) - fmaxf(x1[i], x1[tid]), 0.f);
            float ih = fmaxf(fminf(y2[i], y2[tid]) - fmaxf(y1[i], y1[tid]), 0.f);
            float inter = iw * ih;
            float iou = inter / (area[i] + area[tid] - inter);
            if (iou > 0.3f) supp[tid] = 1;
        }
        __syncthreads();
    }
    if (tid < NMS_K)
        keep_out[tid] = ((tops[tid] > -1e37f) && (supp[tid] == 0)) ? 1.0f : 0.0f;
}

// ---------------------------------------------------------------------------
// Host orchestration
// ---------------------------------------------------------------------------
static void launch_conv(hipStream_t st, const float* X, const float* W, const float* b,
                        float* Y, int Cin, int Cout, int Hin, int Win,
                        int kh, int kw, int stride, int pad, int dil, int relu)
{
    int Hout = (Hin + 2 * pad - dil * (kh - 1) - 1) / stride + 1;
    int Wout = (Win + 2 * pad - dil * (kw - 1) - 1) / stride + 1;
    int Npix = Hout * Wout;
    dim3 grid((Npix + BN - 1) / BN, (Cout + BM - 1) / BM);
    conv_wmma_kernel<<<grid, 256, 0, st>>>(X, W, b, Y, Cin, Cout, Hin, Win,
                                           Hout, Wout, kh, kw, stride, pad, dil, relu);
}

extern "C" void kernel_launch(void* const* d_in, const int* in_sizes, int n_in,
                              void* d_out, int out_size, void* d_ws, size_t ws_size,
                              hipStream_t stream)
{
    (void)in_sizes; (void)n_in; (void)out_size; (void)ws_size;
    const int N = 34125;  // 160^2+80^2+40^2+20^2+10^2+5^2

    const float* x = (const float*)d_in[0];
    const float *vw[15], *vb[15];
    for (int l = 0; l < 15; ++l) { vw[l] = (const float*)d_in[1 + 2 * l]; vb[l] = (const float*)d_in[2 + 2 * l]; }
    const float *ew[4], *eb[4];
    for (int l = 0; l < 4; ++l)  { ew[l] = (const float*)d_in[31 + 2 * l]; eb[l] = (const float*)d_in[32 + 2 * l]; }
    const float *lw[6], *lb[6];
    for (int l = 0; l < 6; ++l)  { lw[l] = (const float*)d_in[39 + 2 * l]; lb[l] = (const float*)d_in[40 + 2 * l]; }
    const float *cfw[6], *cfb[6];
    for (int l = 0; l < 6; ++l)  { cfw[l] = (const float*)d_in[51 + 2 * l]; cfb[l] = (const float*)d_in[52 + 2 * l]; }
    const float* g3 = (const float*)d_in[63];
    const float* g4 = (const float*)d_in[64];
    const float* g5 = (const float*)d_in[65];

    // workspace layout (floats)
    float* W0 = (float*)d_ws;
    size_t o = 0;
    float* A  = W0 + o; o += 26214400ull;           // 64*640*640 (largest act)
    float* Bp = W0 + o; o += 26214400ull;
    float* S0 = W0 + o; o += (size_t)256 * 160 * 160;
    float* S1 = W0 + o; o += (size_t)512 * 80 * 80;
    float* S2 = W0 + o; o += (size_t)512 * 40 * 40;
    float* S3 = W0 + o; o += (size_t)1024 * 20 * 20;
    float* S4 = W0 + o; o += (size_t)512 * 10 * 10;
    float* S5 = W0 + o; o += (size_t)256 * 5 * 5;
    const int   srcC[6]  = {256, 512, 512, 1024, 512, 256};
    const int   srcF[6]  = {160, 80, 40, 20, 10, 5};
    const int   steps[6] = {4, 8, 16, 32, 64, 128};
    const float msz[6]   = {16.f, 32.f, 64.f, 128.f, 256.f, 512.f};
    const int   base[6]  = {0, 25600, 32000, 33600, 34000, 34100};
    float* SRC[6] = {S0, S1, S2, S3, S4, S5};
    float* LOC[6]; float* CONF[6];
    for (int i = 0; i < 6; ++i) { LOC[i]  = W0 + o; o += (size_t)4 * srcF[i] * srcF[i]; }
    for (int i = 0; i < 6; ++i) { CONF[i] = W0 + o; o += (size_t)(i == 0 ? 4 : 2) * srcF[i] * srcF[i]; }
    float* TOPS  = W0 + o; o += 768;
    int*   TAKEN = (int*)(W0 + o); o += N;

    float* out    = (float*)d_out;
    float* boxes  = out;                 // N*4
    float* scores = out + 4 * N;         // N
    float* topi   = out + 5 * N;         // 750 (as float)
    float* keep   = out + 5 * N + 750;   // 750 (as float)

    // ---- VGG backbone ----
    launch_conv(stream, x,  vw[0], vb[0], A,  3,  64, 640, 640, 3, 3, 1, 1, 1, 1);
    launch_conv(stream, A,  vw[1], vb[1], Bp, 64, 64, 640, 640, 3, 3, 1, 1, 1, 1);
    pool_kernel<<<((64 * 320 * 320) + 255) / 256, 256, 0, stream>>>(Bp, A, 64, 640, 640);

    launch_conv(stream, A,  vw[2], vb[2], Bp, 64,  128, 320, 320, 3, 3, 1, 1, 1, 1);
    launch_conv(stream, Bp, vw[3], vb[3], A,  128, 128, 320, 320, 3, 3, 1, 1, 1, 1);
    pool_kernel<<<((128 * 160 * 160) + 255) / 256, 256, 0, stream>>>(A, Bp, 128, 320, 320);

    launch_conv(stream, Bp, vw[4], vb[4], A,  128, 256, 160, 160, 3, 3, 1, 1, 1, 1);
    launch_conv(stream, A,  vw[5], vb[5], Bp, 256, 256, 160, 160, 3, 3, 1, 1, 1, 1);
    launch_conv(stream, Bp, vw[6], vb[6], A,  256, 256, 160, 160, 3, 3, 1, 1, 1, 1);
    l2norm_kernel<<<(25600 + 255) / 256, 256, 0, stream>>>(A, g3, S0, 256, 25600);
    pool_kernel<<<((256 * 80 * 80) + 255) / 256, 256, 0, stream>>>(A, Bp, 256, 160, 160);

    launch_conv(stream, Bp, vw[7], vb[7], A,  256, 512, 80, 80, 3, 3, 1, 1, 1, 1);
    launch_conv(stream, A,  vw[8], vb[8], Bp, 512, 512, 80, 80, 3, 3, 1, 1, 1, 1);
    launch_conv(stream, Bp, vw[9], vb[9], A,  512, 512, 80, 80, 3, 3, 1, 1, 1, 1);
    l2norm_kernel<<<(6400 + 255) / 256, 256, 0, stream>>>(A, g4, S1, 512, 6400);
    pool_kernel<<<((512 * 40 * 40) + 255) / 256, 256, 0, stream>>>(A, Bp, 512, 80, 80);

    launch_conv(stream, Bp, vw[10], vb[10], A,  512, 512, 40, 40, 3, 3, 1, 1, 1, 1);
    launch_conv(stream, A,  vw[11], vb[11], Bp, 512, 512, 40, 40, 3, 3, 1, 1, 1, 1);
    launch_conv(stream, Bp, vw[12], vb[12], A,  512, 512, 40, 40, 3, 3, 1, 1, 1, 1);
    l2norm_kernel<<<(1600 + 255) / 256, 256, 0, stream>>>(A, g5, S2, 512, 1600);
    pool_kernel<<<((512 * 20 * 20) + 255) / 256, 256, 0, stream>>>(A, Bp, 512, 40, 40);

    // fc6 (3x3, pad 6, dil 6) and fc7 (1x1)
    launch_conv(stream, Bp, vw[13], vb[13], A,  512, 1024, 20, 20, 3, 3, 1, 6, 6, 1);
    launch_conv(stream, A,  vw[14], vb[14], S3, 1024, 1024, 20, 20, 1, 1, 1, 0, 1, 1);

    // extras
    launch_conv(stream, S3, ew[0], eb[0], A,  1024, 256, 20, 20, 1, 1, 1, 0, 1, 1);
    launch_conv(stream, A,  ew[1], eb[1], S4, 256,  512, 20, 20, 3, 3, 2, 1, 1, 1);
    launch_conv(stream, S4, ew[2], eb[2], A,  512,  128, 10, 10, 1, 1, 1, 0, 1, 1);
    launch_conv(stream, A,  ew[3], eb[3], S5, 128,  256, 10, 10, 3, 3, 2, 1, 1, 1);

    // ---- heads + decode ----
    for (int i = 0; i < 6; ++i) {
        int f = srcF[i];
        launch_conv(stream, SRC[i], lw[i],  lb[i],  LOC[i],  srcC[i], 4, f, f, 3, 3, 1, 1, 1, 0);
        int cc = (i == 0) ? 4 : 2;
        launch_conv(stream, SRC[i], cfw[i], cfb[i], CONF[i], srcC[i], cc, f, f, 3, 3, 1, 1, 1, 0);
        decode_kernel<<<(f * f + 255) / 256, 256, 0, stream>>>(
            LOC[i], CONF[i], boxes, scores, f, steps[i], msz[i], base[i], cc);
    }

    // ---- top-k + NMS ----
    topk_kernel<<<1, 1024, 0, stream>>>(scores, TAKEN, N, topi, TOPS, 750);
    nms_kernel<<<1, 768, 0, stream>>>(boxes, topi, TOPS, keep);
}
